// GCNSpatialBlock_39350490366433
// MI455X (gfx1250) — matrix operand order repaired and checked by reading, not compiled, not run on hardware
//
#include <hip/hip_runtime.h>
#include <hip/hip_bf16.h>

typedef __attribute__((ext_vector_type(16))) _Float16 v16h;
typedef __attribute__((ext_vector_type(8)))  _Float16 v8h;
typedef __attribute__((ext_vector_type(8)))  float    v8f;

#define N_B   64
#define C0    64
#define V_B   25
#define T_B   128
#define VT    (V_B * T_B)          // 3200
#define NTOK  (N_B * VT)           // 204800
#define GP    16
#define BN_EPS 1e-5f

// ---------------------------------------------------------------------------
// WMMA tile loaders (CDNA5 16x16x32 f16 layouts, wave32)
// A (16xK slab, row-major W, ld = row stride): lane l -> row l&15,
//   16 contiguous K halves starting at k0 + off,
//   with off = (lane&16) ? 16 : 0.
// B (Kx16 slab from token-major X, ld = token stride): lane l -> col tok0+(l&15),
//   16 contiguous K halves starting at k0 + ((lane&16) ? 16 : 0).
// C/D: VGPR i, lanes 0-15 -> row i, col lane; lanes 16-31 -> row i+8, col lane-16.
// ---------------------------------------------------------------------------
static __device__ __forceinline__ v16h wmma_load_a(const _Float16* __restrict__ W,
                                                   int ld, int row, int k0, int lane) {
  const _Float16* p = W + (size_t)row * ld + k0 + ((lane & 16) ? 16 : 0);
  v8h lo = *(const v8h*)(p);
  v8h hi = *(const v8h*)(p + 8);
  v16h a;
#pragma unroll
  for (int i = 0; i < 8; ++i) { a[i] = lo[i]; a[i + 8] = hi[i]; }
  return a;
}

static __device__ __forceinline__ v16h wmma_load_b(const _Float16* __restrict__ X,
                                                   int ld, int k0, int col, int lane) {
  const _Float16* p = X + (size_t)col * ld + k0 + ((lane & 16) ? 16 : 0);
  return *(const v16h*)(p);
}

// ---------------------------------------------------------------------------
// f32 -> f16 weight conversion
// ---------------------------------------------------------------------------
__global__ void cvt_f32_f16(const float* __restrict__ src, _Float16* __restrict__ dst, int n) {
  int i = blockIdx.x * 256 + threadIdx.x;
  if (i < n) dst[i] = (_Float16)src[i];
}

// ---------------------------------------------------------------------------
// x (N,C,V,T) f32 -> xh token-major [n][v][t][c] f16 via LDS tile transpose
// one block per (n,v): transposes the 64(c) x 128(t) slab
// ---------------------------------------------------------------------------
__global__ __launch_bounds__(256)
void x_to_tok16(const float* __restrict__ X, _Float16* __restrict__ Xh) {
  __shared__ _Float16 tile[C0][T_B + 8];
  int nv = blockIdx.x;                 // n*V_B + v
  int n = nv / V_B, v = nv % V_B;
#pragma unroll
  for (int j = 0; j < 8; ++j) {
    int idx = threadIdx.x + j * 256;   // 2048 = 16*128
    int c = idx >> 7, t = idx & 127;
    float val = X[((size_t)(n * C0 + c) * V_B + v) * T_B + t];
    tile[c][t] = (_Float16)val;
  }
  __syncthreads();
#pragma unroll
  for (int j = 0; j < 4; ++j) {
    int id = threadIdx.x + j * 256;    // 1024 = 128 tokens * 8 chunks
    int t = id >> 3, oc = (id & 7) * 8;
    v8h w;
#pragma unroll
    for (int i = 0; i < 8; ++i) w[i] = tile[oc + i][t];
    *(v8h*)(Xh + ((size_t)(n * V_B + v) * T_B + t) * C0 + oc) = w;
  }
}

// ---------------------------------------------------------------------------
// a1/a2 = wg1,wg2 (16x64) @ X (64 x NTOK) via WMMA; outputs stored transposed
// as [n][t][v][16] f16 for the per-(n,t) adjacency build.
// one wave per 16-token tile (12800 waves)
// ---------------------------------------------------------------------------
__global__ __launch_bounds__(256)
void gp_gemm(const _Float16* __restrict__ Wg1, const _Float16* __restrict__ Wg2,
             const _Float16* __restrict__ X,
             _Float16* __restrict__ A1t, _Float16* __restrict__ A2t) {
  int lane = threadIdx.x & 31;
  int wave = blockIdx.x * 8 + (threadIdx.x >> 5);
  int tok0 = wave << 4;
  v8f c1 = {0.f, 0.f, 0.f, 0.f, 0.f, 0.f, 0.f, 0.f};
  v8f c2 = c1;
  int row = lane & 15;
#pragma unroll
  for (int k0 = 0; k0 < C0; k0 += 32) {
    v16h a1 = wmma_load_a(Wg1, C0, row, k0, lane);
    v16h a2 = wmma_load_a(Wg2, C0, row, k0, lane);
    v16h b  = wmma_load_b(X, C0, k0, tok0 + (lane & 15), lane);
    c1 = __builtin_amdgcn_wmma_f32_16x16x32_f16(false, a1, false, b, (short)0, c1, false, false);
    c2 = __builtin_amdgcn_wmma_f32_16x16x32_f16(false, a2, false, b, (short)0, c2, false, false);
  }
  // lane holds token col = tok0+(lane&15), rows (lane<16 ? 0..7 : 8..15)
  int tok = tok0 + (lane & 15);
  int n = tok / VT; int rem = tok % VT; int v = rem / T_B; int t = rem % T_B;
  size_t obase = ((size_t)(n * T_B + t) * V_B + v) * GP + ((lane & 16) ? 8 : 0);
  v8h h1, h2;
#pragma unroll
  for (int i = 0; i < 8; ++i) { h1[i] = (_Float16)c1[i]; h2[i] = (_Float16)c2[i]; }
  *(v8h*)(A1t + obase) = h1;
  *(v8h*)(A2t + obase) = h2;
}

// ---------------------------------------------------------------------------
// g[n,t,v,u] = softmax_u( sum_o a1[o,v]*a2[o,u] ) ; one block per (n,t)
// (0.16 GFLOP total -> VALU; avoids V=25 padding gymnastics)
// ---------------------------------------------------------------------------
__global__ __launch_bounds__(256)
void g_softmax(const _Float16* __restrict__ A1t, const _Float16* __restrict__ A2t,
               float* __restrict__ G) {
  __shared__ float a1[V_B][GP], a2[V_B][GP], sc[V_B][V_B + 1];
  int nt = blockIdx.x;                 // n*T_B + t
  const _Float16* p1 = A1t + (size_t)nt * V_B * GP;
  const _Float16* p2 = A2t + (size_t)nt * V_B * GP;
  for (int i = threadIdx.x; i < V_B * GP; i += 256) {
    a1[i / GP][i % GP] = (float)p1[i];
    a2[i / GP][i % GP] = (float)p2[i];
  }
  __syncthreads();
  for (int i = threadIdx.x; i < V_B * V_B; i += 256) {
    int v = i / V_B, u = i % V_B;
    float s = 0.f;
#pragma unroll
    for (int k = 0; k < GP; ++k) s += a1[v][k] * a2[u][k];
    sc[v][u] = s;
  }
  __syncthreads();
  if (threadIdx.x < V_B) {
    int v = threadIdx.x;
    float m = -3.4e38f;
    for (int u = 0; u < V_B; ++u) m = fmaxf(m, sc[v][u]);
    float sum = 0.f;
    for (int u = 0; u < V_B; ++u) { float e = __expf(sc[v][u] - m); sc[v][u] = e; sum += e; }
    float inv = 1.f / sum;
    float* gout = G + (size_t)nt * V_B * V_B + (size_t)v * V_B;
    for (int u = 0; u < V_B; ++u) gout[u] = sc[v][u] * inv;
  }
}

// ---------------------------------------------------------------------------
// xm[n,v,t,:] = sum_u g[n,t,v,u] * h[n,u,t,:]   (one block per (n,t), LDS-resident)
// ---------------------------------------------------------------------------
__global__ __launch_bounds__(256)
void mix_kernel(const float* __restrict__ G, const _Float16* __restrict__ H,
                _Float16* __restrict__ XM, int Cin) {
  __shared__ float gs[V_B][V_B + 1];
  __shared__ float hsf[V_B * 128];
  int nt = blockIdx.x;
  int n = nt / T_B, t = nt % T_B;
  const float* gp = G + (size_t)nt * V_B * V_B;
  for (int i = threadIdx.x; i < V_B * V_B; i += 256) gs[i / V_B][i % V_B] = gp[i];
  for (int i = threadIdx.x; i < V_B * Cin; i += 256) {
    int u = i / Cin, c = i % Cin;
    hsf[u * Cin + c] = (float)H[((size_t)(n * V_B + u) * T_B + t) * Cin + c];
  }
  __syncthreads();
  for (int i = threadIdx.x; i < V_B * Cin; i += 256) {
    int v = i / Cin, c = i % Cin;
    float s = 0.f;
#pragma unroll 5
    for (int u = 0; u < V_B; ++u) s += gs[v][u] * hsf[u * Cin + c];
    XM[((size_t)(n * V_B + v) * T_B + t) * Cin + c] = (_Float16)s;
  }
}

// ---------------------------------------------------------------------------
// Dual WMMA GEMM: Y(O x NTOK) = W1(OxCin)@XM + W2(OxCin)@H, f16 in / f32 out.
// Each wave computes a 32(O) x 32(token) block: 2x2 accumulator tiles with
// 2x reuse on both A and B slabs; both products accumulate into the same f32
// tiles. The 4 W1*X1 WMMAs issue before the dependent W2*X2 accumulations so
// each RAW accumulate chain is separated by 3 independent WMMAs.
// ---------------------------------------------------------------------------
template <int CIN>
__global__ __launch_bounds__(256)
void gemm_dual_f16(const _Float16* __restrict__ W1, const _Float16* __restrict__ W2,
                   const _Float16* __restrict__ X1, const _Float16* __restrict__ X2,
                   float* __restrict__ Y, int O) {
  int lane = threadIdx.x & 31;
  int wave = blockIdx.x * 8 + (threadIdx.x >> 5);
  int mblocks = O >> 5;                // 32-row blocks
  int o0  = (wave % mblocks) << 5;
  int tb0 = (wave / mblocks) << 5;     // 32-token blocks
  v8f acc[2][2];
#pragma unroll
  for (int m = 0; m < 2; ++m)
#pragma unroll
    for (int j = 0; j < 2; ++j)
      acc[m][j] = (v8f){0.f, 0.f, 0.f, 0.f, 0.f, 0.f, 0.f, 0.f};
  int r = lane & 15;
#pragma unroll
  for (int k0 = 0; k0 < CIN; k0 += 32) {
    v16h a1[2], a2[2], b1[2], b2[2];
#pragma unroll
    for (int m = 0; m < 2; ++m) {
      a1[m] = wmma_load_a(W1, CIN, o0 + m * 16 + r, k0, lane);
      a2[m] = wmma_load_a(W2, CIN, o0 + m * 16 + r, k0, lane);
    }
#pragma unroll
    for (int j = 0; j < 2; ++j) {
      b1[j] = wmma_load_b(X1, CIN, k0, tb0 + j * 16 + r, lane);
      b2[j] = wmma_load_b(X2, CIN, k0, tb0 + j * 16 + r, lane);
    }
#pragma unroll
    for (int m = 0; m < 2; ++m)
#pragma unroll
      for (int j = 0; j < 2; ++j)
        acc[m][j] = __builtin_amdgcn_wmma_f32_16x16x32_f16(false, a1[m], false, b1[j],
                                                           (short)0, acc[m][j], false, false);
#pragma unroll
    for (int m = 0; m < 2; ++m)
#pragma unroll
      for (int j = 0; j < 2; ++j)
        acc[m][j] = __builtin_amdgcn_wmma_f32_16x16x32_f16(false, a2[m], false, b2[j],
                                                           (short)0, acc[m][j], false, false);
  }
#pragma unroll
  for (int m = 0; m < 2; ++m) {
    int rbase = o0 + m * 16 + ((lane & 16) ? 8 : 0);
#pragma unroll
    for (int j = 0; j < 2; ++j) {
      int col = tb0 + j * 16 + r;
#pragma unroll
      for (int i = 0; i < 8; ++i)
        Y[(size_t)(rbase + i) * NTOK + col] = acc[m][j][i];
    }
  }
}

// ---------------------------------------------------------------------------
// BN stats: one block per channel, deterministic tree reduction (no atomics)
// ---------------------------------------------------------------------------
__global__ __launch_bounds__(256)
void bn_stats(const float* __restrict__ Y, float* __restrict__ st, int O) {
  int o = blockIdx.x;
  const float* p = Y + (size_t)o * NTOK;
  float s0 = 0.f, s1 = 0.f, q0 = 0.f, q1 = 0.f;
  for (int i = threadIdx.x; i < NTOK; i += 512) {   // NTOK % 512 == 0
    float a = p[i], b = p[i + 256];
    s0 += a; q0 += a * a; s1 += b; q1 += b * b;
  }
  __shared__ float rs[256], rq[256];
  rs[threadIdx.x] = s0 + s1;
  rq[threadIdx.x] = q0 + q1;
  __syncthreads();
  for (int k = 128; k > 0; k >>= 1) {
    if ((int)threadIdx.x < k) { rs[threadIdx.x] += rs[threadIdx.x + k]; rq[threadIdx.x] += rq[threadIdx.x + k]; }
    __syncthreads();
  }
  if (threadIdx.x == 0) { st[o] = rs[0]; st[O + o] = rq[0]; }
}

__global__ void bn_finalize(const float* __restrict__ st, const float* __restrict__ gamma,
                            const float* __restrict__ beta, float* __restrict__ scsh, int O) {
  int o = blockIdx.x * 256 + threadIdx.x;
  if (o < O) {
    float inv_n = 1.0f / (float)NTOK;
    float mean = st[o] * inv_n;
    float var  = st[O + o] * inv_n - mean * mean;
    float sc = gamma[o] * rsqrtf(var + BN_EPS);
    scsh[o] = sc;
    scsh[O + o] = beta[o] - mean * sc;
  }
}

// ---------------------------------------------------------------------------
// BN+ReLU epilogue -> next-stage f16 token-major input (LDS tile transpose)
// ---------------------------------------------------------------------------
__global__ __launch_bounds__(256)
void bn_relu_t16(const float* __restrict__ Y, const float* __restrict__ scsh,
                 _Float16* __restrict__ Hn, int O) {
  __shared__ _Float16 tile[32][72];
  int o0 = blockIdx.y * 32, t0 = blockIdx.x * 64;
#pragma unroll
  for (int j = 0; j < 8; ++j) {
    int idx = threadIdx.x + j * 256;   // 2048 = 32*64
    int ol = idx >> 6, tl = idx & 63;
    float v = Y[(size_t)(o0 + ol) * NTOK + t0 + tl];
    float r = v * scsh[o0 + ol] + scsh[O + o0 + ol];
    tile[ol][tl] = (_Float16)fmaxf(r, 0.f);
  }
  __syncthreads();
  int tok = threadIdx.x >> 2, oc = (threadIdx.x & 3) * 8;
  v8h w;
#pragma unroll
  for (int i = 0; i < 8; ++i) w[i] = tile[oc + i][tok];
  *(v8h*)(Hn + (size_t)(t0 + tok) * O + o0 + oc) = w;
}

// ---------------------------------------------------------------------------
// Final-stage BN+ReLU -> d_out f32 in (N,256,V,T) layout
// ---------------------------------------------------------------------------
__global__ __launch_bounds__(256)
void bn_relu_out(const float* __restrict__ Y, const float* __restrict__ scsh,
                 float* __restrict__ Out) {
  size_t i = (size_t)blockIdx.x * 256 + threadIdx.x;   // [o][n][vt] order
  int r = (int)(i % VT);
  size_t q = i / VT;
  int n = (int)(q % N_B);
  int o = (int)(q / N_B);
  float v = Y[i] * scsh[o] + scsh[256 + o];
  Out[((size_t)(n * 256 + o)) * VT + r] = fmaxf(v, 0.f);
}

// ---------------------------------------------------------------------------
extern "C" void kernel_launch(void* const* d_in, const int* in_sizes, int n_in,
                              void* d_out, int out_size, void* d_ws, size_t ws_size,
                              hipStream_t stream) {
  (void)in_sizes; (void)n_in; (void)out_size; (void)ws_size;
  const float* x      = (const float*)d_in[0];
  const float* wg1    = (const float*)d_in[1];
  const float* wg2    = (const float*)d_in[2];
  const float* w11    = (const float*)d_in[3];
  const float* w12    = (const float*)d_in[4];
  const float* gamma1 = (const float*)d_in[5];
  const float* beta1  = (const float*)d_in[6];
  const float* w21    = (const float*)d_in[7];
  const float* w22    = (const float*)d_in[8];
  const float* gamma2 = (const float*)d_in[9];
  const float* beta2  = (const float*)d_in[10];
  const float* w31    = (const float*)d_in[11];
  const float* w32    = (const float*)d_in[12];
  const float* gamma3 = (const float*)d_in[13];
  const float* beta3  = (const float*)d_in[14];

  float* xout = (float*)d_out;                           // (N,256,V,T)
  float* gout = (float*)d_out + (size_t)N_B * 256 * VT;  // (N,T,V,V)

  char* ws = (char*)d_ws;
  size_t off = 0;
  auto take = [&](size_t bytes) -> char* {
    char* p = ws + off;
    off = (off + bytes + 255) & ~(size_t)255;
    return p;
  };
  _Float16* wg1h = (_Float16*)take((size_t)GP * C0 * 2);
  _Float16* wg2h = (_Float16*)take((size_t)GP * C0 * 2);
  _Float16* w11h = (_Float16*)take((size_t)64 * 64 * 2);
  _Float16* w12h = (_Float16*)take((size_t)64 * 64 * 2);
  _Float16* w21h = (_Float16*)take((size_t)128 * 64 * 2);
  _Float16* w22h = (_Float16*)take((size_t)128 * 64 * 2);
  _Float16* w31h = (_Float16*)take((size_t)256 * 128 * 2);
  _Float16* w32h = (_Float16*)take((size_t)256 * 128 * 2);
  _Float16* xh   = (_Float16*)take((size_t)NTOK * 64 * 2);
  _Float16* a1t  = (_Float16*)take((size_t)NTOK * GP * 2);
  _Float16* a2t  = (_Float16*)take((size_t)NTOK * GP * 2);
  _Float16* hA   = (_Float16*)take((size_t)NTOK * 64 * 2);
  _Float16* hB   = (_Float16*)take((size_t)NTOK * 128 * 2);
  _Float16* xm   = (_Float16*)take((size_t)NTOK * 128 * 2);
  float*    ybuf = (float*)take((size_t)NTOK * 256 * 4);
  float*    stats = (float*)take(512 * 4);
  float*    scsh  = (float*)take(512 * 4);

  // --- weight conversion ---
  cvt_f32_f16<<<4, 256, 0, stream>>>(wg1, wg1h, GP * C0);
  cvt_f32_f16<<<4, 256, 0, stream>>>(wg2, wg2h, GP * C0);
  cvt_f32_f16<<<16, 256, 0, stream>>>(w11, w11h, 64 * 64);
  cvt_f32_f16<<<16, 256, 0, stream>>>(w12, w12h, 64 * 64);
  cvt_f32_f16<<<32, 256, 0, stream>>>(w21, w21h, 128 * 64);
  cvt_f32_f16<<<32, 256, 0, stream>>>(w22, w22h, 128 * 64);
  cvt_f32_f16<<<128, 256, 0, stream>>>(w31, w31h, 256 * 128);
  cvt_f32_f16<<<128, 256, 0, stream>>>(w32, w32h, 256 * 128);

  // --- x -> f16 token-major ---
  x_to_tok16<<<N_B * V_B, 256, 0, stream>>>(x, xh);

  // --- adjacency: a1/a2 GEMM (WMMA) + softmax ---
  gp_gemm<<<NTOK / 16 / 8, 256, 0, stream>>>(wg1h, wg2h, xh, a1t, a2t);
  g_softmax<<<N_B * T_B, 256, 0, stream>>>(a1t, a2t, gout);

  // --- stage 1: 64 -> 64 ---
  mix_kernel<<<N_B * T_B, 256, 0, stream>>>(gout, xh, xm, 64);
  gemm_dual_f16<64><<<(64 / 32) * (NTOK / 32) / 8, 256, 0, stream>>>(w11h, w12h, xm, xh, ybuf, 64);
  bn_stats<<<64, 256, 0, stream>>>(ybuf, stats, 64);
  bn_finalize<<<1, 256, 0, stream>>>(stats, gamma1, beta1, scsh, 64);
  bn_relu_t16<<<dim3(NTOK / 64, 64 / 32), 256, 0, stream>>>(ybuf, scsh, hA, 64);

  // --- stage 2: 64 -> 128 ---
  mix_kernel<<<N_B * T_B, 256, 0, stream>>>(gout, hA, xm, 64);
  gemm_dual_f16<64><<<(128 / 32) * (NTOK / 32) / 8, 256, 0, stream>>>(w21h, w22h, xm, hA, ybuf, 128);
  bn_stats<<<128, 256, 0, stream>>>(ybuf, stats, 128);
  bn_finalize<<<1, 256, 0, stream>>>(stats, gamma2, beta2, scsh, 128);
  bn_relu_t16<<<dim3(NTOK / 64, 128 / 32), 256, 0, stream>>>(ybuf, scsh, hB, 128);

  // --- stage 3: 128 -> 256 ---
  mix_kernel<<<N_B * T_B, 256, 0, stream>>>(gout, hB, xm, 128);
  gemm_dual_f16<128><<<(256 / 32) * (NTOK / 32) / 8, 256, 0, stream>>>(w31h, w32h, xm, hB, ybuf, 256);
  bn_stats<<<256, 256, 0, stream>>>(ybuf, stats, 256);
  bn_finalize<<<1, 256, 0, stream>>>(stats, gamma3, beta3, scsh, 256);
  bn_relu_out<<<(int)(((size_t)256 * NTOK) / 256), 256, 0, stream>>>(ybuf, scsh, xout);
}
